// MultiHeadAttention_68874095558886
// MI455X (gfx1250) — matrix operand order repaired and checked by reading, not compiled, not run on hardware
//
#include <hip/hip_runtime.h>

// ---------------------------------------------------------------------------
// MHA forward for MI455X (gfx1250, wave32, WMMA).
// Pipeline: cvt(fp32->bf16, weights transposed) -> GEMM(qkv) -> flash attn
//           -> GEMM(proj).
// All matrix math on v_wmma_f32_16x16x32_bf16 with fp32 accumulation.
// Weights are pre-transposed to [N][K] so both A and B WMMA fragments read
// contraction-contiguous bf16 pairs: no interleave VALU in the GEMM hot loop.
// ---------------------------------------------------------------------------

typedef __attribute__((ext_vector_type(16))) __bf16 v16bf;
typedef __attribute__((ext_vector_type(8)))  float  v8f;

union FragBF16 {
    v16bf v;
    unsigned int u[8];
};

static __device__ __forceinline__ v8f wmma_bf16(v16bf a, v16bf b, v8f c) {
    // args: (neg_a, A, neg_b, B, c_mod, C, reuse_a, reuse_b)
    return __builtin_amdgcn_wmma_f32_16x16x32_bf16(false, a, false, b, (short)0, c, false, false);
}

static __device__ __forceinline__ v8f v8f_zero() {
    v8f z;
#pragma unroll
    for (int i = 0; i < 8; ++i) z[i] = 0.0f;
    return z;
}

// fp32 -> bf16 round-to-nearest-even, as raw u16
static __device__ __forceinline__ unsigned short f2bf(float f) {
    unsigned int u = __float_as_uint(f);
    unsigned int r = u + 0x7fffu + ((u >> 16) & 1u);
    return (unsigned short)(r >> 16);
}

// ---------------------------------------------------------------------------
// Stage 0a: fp32 -> bf16 linear conversion (activations; K-contiguous already)
// ---------------------------------------------------------------------------
__global__ __launch_bounds__(256) void cvt_bf16_kernel(const float* __restrict__ src,
                                                       unsigned short* __restrict__ dst,
                                                       int n4) {
    int i = blockIdx.x * blockDim.x + threadIdx.x;
    if (i >= n4) return;
    float4 f = ((const float4*)src)[i];
    unsigned long long w = (unsigned long long)f2bf(f.x)
                         | ((unsigned long long)f2bf(f.y) << 16)
                         | ((unsigned long long)f2bf(f.z) << 32)
                         | ((unsigned long long)f2bf(f.w) << 48);
    ((unsigned long long*)dst)[i] = w;
}

// ---------------------------------------------------------------------------
// Stage 0b: fp32 [K][N] -> bf16 [N][K] transposed conversion (weights).
// 32x32 LDS-tiled transpose, coalesced global reads and writes.
// ---------------------------------------------------------------------------
__global__ __launch_bounds__(256) void cvt_transpose_kernel(const float* __restrict__ src,
                                                            unsigned short* __restrict__ dst,
                                                            int K, int N) {
    __shared__ float tile[32][33];
    const int k0 = blockIdx.y * 32;
    const int n0 = blockIdx.x * 32;
    {
        const int c = threadIdx.x & 31;
        const int r = threadIdx.x >> 5;  // 0..7
#pragma unroll
        for (int i = 0; i < 4; ++i) {
            const int k = r + i * 8;
            tile[k][c] = src[(size_t)(k0 + k) * N + n0 + c];
        }
    }
    __syncthreads();
    {
        const int n  = threadIdx.x >> 3;        // 0..31
        const int kc = (threadIdx.x & 7) * 4;   // 0..28
        unsigned long long w = (unsigned long long)f2bf(tile[kc + 0][n])
                             | ((unsigned long long)f2bf(tile[kc + 1][n]) << 16)
                             | ((unsigned long long)f2bf(tile[kc + 2][n]) << 32)
                             | ((unsigned long long)f2bf(tile[kc + 3][n]) << 48);
        *(unsigned long long*)&dst[(size_t)(n0 + n) * K + k0 + kc] = w;
    }
}

// ---------------------------------------------------------------------------
// Stage 1/3: tiled bf16 GEMM. BM=128, BN=256, BK=32, 256 threads = 8 waves.
// Wave grid 2(M) x 4(N); each wave: 4x4 WMMA tiles of 16x16 (64x64).
// Double-buffered LDS, single barrier per k-step, B from transposed weights.
//   mode 0: C = A @ Wqkv + b_qkv -> Q[B,H,T,D] (scaled 1/sqrt(D)),
//           K[B,H,T,D], Vt[B,H,D,T]  (all bf16)
//   mode 1: C = A @ Wproj + b_proj -> fp32 d_out [M][2048]
// A:  [8192][2048] bf16 row-major (k contiguous)
// Bt: [N][2048]    bf16 row-major (k contiguous; transposed weights)
// ---------------------------------------------------------------------------
__global__ __launch_bounds__(256, 1) void gemm_bf16_kernel(
    const unsigned short* __restrict__ A,
    const unsigned short* __restrict__ Bt,
    const float* __restrict__ bias,
    int mode,
    unsigned short* __restrict__ qOut,
    unsigned short* __restrict__ kOut,
    unsigned short* __restrict__ vtOut,
    float* __restrict__ fOut) {

    const int t  = threadIdx.x;
    const int bm = blockIdx.y * 128;
    const int bn = blockIdx.x * 256;

    __shared__ alignas(16) unsigned short As[2][128][32];  // [m][k]
    __shared__ alignas(16) unsigned short Bs[2][256][32];  // [n][k]

    const int wave  = t >> 5;
    const int lane  = t & 31;
    const int l15   = lane & 15;
    const int lhalf = (lane & 16) ? 1 : 0;
    const int wm    = (wave & 1) * 64;   // 2 waves along M
    const int wn    = (wave >> 1) * 64;  // 4 waves along N

    // staging assignments
    const int arow = t >> 1;             // A: 128 rows, 2 threads/row (32B each)
    const int acol = (t & 1) * 16;

    const unsigned short* agBase = A  + (size_t)(bm + arow) * 2048 + acol;
    const unsigned short* bgBase = Bt + (size_t)(bn + t) * 2048;  // one n-row/thread

    v8f acc[4][4];
#pragma unroll
    for (int i = 0; i < 4; ++i)
#pragma unroll
        for (int j = 0; j < 4; ++j) acc[i][j] = v8f_zero();

    const int kb  = lhalf ? 8 : 0;    // A-frag K base
    const int jb2 = lhalf ? 8 : 0;    // B-frag K base

    // ---- prologue: stage k-step 0 into buffer 0 ----
    {
        uint4 a0 = *(const uint4*)agBase;
        uint4 a1 = *(const uint4*)(agBase + 8);
        *(uint4*)&As[0][arow][acol]     = a0;
        *(uint4*)&As[0][arow][acol + 8] = a1;
        uint4 b0 = *(const uint4*)bgBase;
        uint4 b1 = *(const uint4*)(bgBase + 8);
        uint4 b2 = *(const uint4*)(bgBase + 16);
        uint4 b3 = *(const uint4*)(bgBase + 24);
        *(uint4*)&Bs[0][t][0]  = b0;
        *(uint4*)&Bs[0][t][8]  = b1;
        *(uint4*)&Bs[0][t][16] = b2;
        *(uint4*)&Bs[0][t][24] = b3;
    }
    __syncthreads();

    for (int kt = 0; kt < 64; ++kt) {
        const int c = kt & 1;
        const bool hasNext = (kt < 63);

        // ---- issue global loads for next k-step (hidden behind WMMAs) ----
        uint4 nA0, nA1, nB0, nB1, nB2, nB3;
        if (hasNext) {
            const int k1 = (kt + 1) * 32;
            nA0 = *(const uint4*)(agBase + k1);
            nA1 = *(const uint4*)(agBase + k1 + 8);
            nB0 = *(const uint4*)(bgBase + k1);
            nB1 = *(const uint4*)(bgBase + k1 + 8);
            nB2 = *(const uint4*)(bgBase + k1 + 16);
            nB3 = *(const uint4*)(bgBase + k1 + 24);
            if (kt < 62) {
                __builtin_prefetch(agBase + (kt + 2) * 32, 0, 1);
                __builtin_prefetch(bgBase + (kt + 2) * 32, 0, 1);
            }
        }

        // ---- fragments from current buffer (contiguous bf16 pairs) ----
        FragBF16 fa[4];
#pragma unroll
        for (int mt = 0; mt < 4; ++mt) {
            const int mr = wm + mt * 16 + l15;
#pragma unroll
            for (int j = 0; j < 8; ++j) {
                const int kk = (j < 4) ? (kb + 2 * j) : (kb + 16 + 2 * (j - 4));
                fa[mt].u[j] = *(const unsigned int*)&As[c][mr][kk];
            }
        }
        FragBF16 fb[4];
#pragma unroll
        for (int nt = 0; nt < 4; ++nt) {
            const int nc = wn + nt * 16 + l15;
#pragma unroll
            for (int j = 0; j < 8; ++j)
                fb[nt].u[j] = *(const unsigned int*)&Bs[c][nc][jb2 + 2 * j + ((j < 4) ? 0 : 8)];
        }
#pragma unroll
        for (int mt = 0; mt < 4; ++mt)
#pragma unroll
            for (int nt = 0; nt < 4; ++nt)
                acc[mt][nt] = wmma_bf16(fa[mt].v, fb[nt].v, acc[mt][nt]);

        // ---- store next tile into the other buffer ----
        if (hasNext) {
            const int nb = 1 - c;
            *(uint4*)&As[nb][arow][acol]     = nA0;
            *(uint4*)&As[nb][arow][acol + 8] = nA1;
            *(uint4*)&Bs[nb][t][0]  = nB0;
            *(uint4*)&Bs[nb][t][8]  = nB1;
            *(uint4*)&Bs[nb][t][16] = nB2;
            *(uint4*)&Bs[nb][t][24] = nB3;
        }
        __syncthreads();
    }

    // ---- epilogue ----
    const float qscale = 0.08838834764831845f; // 1/sqrt(128)
#pragma unroll
    for (int mt = 0; mt < 4; ++mt) {
#pragma unroll
        for (int nt = 0; nt < 4; ++nt) {
#pragma unroll
            for (int r = 0; r < 8; ++r) {
                const int row = bm + wm + mt * 16 + r + lhalf * 8;
                const int col = bn + wn + nt * 16 + l15;
                const float v = acc[mt][nt][r] + bias[col];
                if (mode == 0) {
                    const int which = col >> 11;
                    const int cc = col & 2047;
                    const int h = cc >> 7;
                    const int d = cc & 127;
                    const int b = row >> 11;
                    const int tp = row & 2047;
                    const int bh = b * 16 + h;
                    if (which == 0)
                        qOut[((size_t)bh * 2048 + tp) * 128 + d] = f2bf(v * qscale);
                    else if (which == 1)
                        kOut[((size_t)bh * 2048 + tp) * 128 + d] = f2bf(v);
                    else
                        vtOut[((size_t)bh * 128 + d) * 2048 + tp] = f2bf(v);
                } else {
                    fOut[(size_t)row * 2048 + col] = v;
                }
            }
        }
    }
}

// ---------------------------------------------------------------------------
// Stage 2: causal flash attention.
// Grid (T/64, B*H); 128 threads = 4 waves; each wave owns 16 query rows.
// Block-cooperative LDS staging of K (32x128) and V^T (128x32) tiles,
// block-uniform kv trip count (fully-masked blocks are numerically inert).
// Online softmax with half-group shfl_xor reductions; P re-layout C->A via
// per-wave LDS tile. Q pre-scaled by 1/sqrt(D).
// ---------------------------------------------------------------------------
__global__ __launch_bounds__(128) void flash_attn_kernel(
    const unsigned short* __restrict__ Q,   // [B,H,T,D] bf16 (scaled)
    const unsigned short* __restrict__ K,   // [B,H,T,D] bf16
    const unsigned short* __restrict__ Vt,  // [B,H,D,T] bf16
    unsigned short* __restrict__ O) {       // [B*T, C]  bf16

    const int t     = threadIdx.x;
    const int wave  = t >> 5;
    const int lane  = t & 31;
    const int l15   = lane & 15;
    const int lhalf = (lane & 16) ? 1 : 0;
    const int bh    = blockIdx.y;
    const int qbase = blockIdx.x * 64 + wave * 16;

    const unsigned short* Qb = Q + (size_t)bh * 2048 * 128;
    const unsigned short* Kb = K + (size_t)bh * 2048 * 128;
    const unsigned short* Vb = Vt + (size_t)bh * 128 * 2048;

    __shared__ alignas(16) unsigned short Ks[32][128];     // [kv_local][d]
    __shared__ alignas(16) unsigned short Vs[128][32];     // [d][kv_local]
    __shared__ alignas(16) unsigned short Pls[4][16][32];  // per-wave P tile

    // staging assignments (128 threads, 8KB each tile => 64B per thread)
    const int krow = t >> 2;            // 0..31
    const int kcol = (t & 3) * 32;      // 0,32,64,96
    const int vrow = t;                 // 0..127 (d)

    // Q fragments for 4 chunks of 32 along D, kept in registers
    FragBF16 qf[4];
    {
        const int mr = qbase + l15;
        const int kbq = lhalf ? 8 : 0;
#pragma unroll
        for (int kk = 0; kk < 4; ++kk) {
#pragma unroll
            for (int j = 0; j < 8; ++j) {
                const int d = kk * 32 + ((j < 4) ? (kbq + 2 * j) : (kbq + 16 + 2 * (j - 4)));
                qf[kk].u[j] = *(const unsigned int*)(Qb + (size_t)mr * 128 + d);
            }
        }
    }

    v8f o[8];
#pragma unroll
    for (int i = 0; i < 8; ++i) o[i] = v8f_zero();
    float Mrun[8], Lrun[8];
#pragma unroll
    for (int r = 0; r < 8; ++r) { Mrun[r] = -1e30f; Lrun[r] = 0.0f; }

    const int db  = lhalf ? 8 : 0;   // K-frag contraction base (over d)
    const int tb  = lhalf ? 8 : 0;   // V-frag contraction base (over kv)
    const int kb2 = lhalf ? 8 : 0;   // P A-frag K base

    const int kvEnd = (blockIdx.x + 1) * 64;  // block-uniform causal bound

    for (int kv = 0; kv < kvEnd; kv += 32) {
        // ---- cooperative staging of K and V tiles ----
        {
            const unsigned short* kg = Kb + (size_t)(kv + krow) * 128 + kcol;
            uint4 k0 = *(const uint4*)kg;
            uint4 k1 = *(const uint4*)(kg + 8);
            uint4 k2 = *(const uint4*)(kg + 16);
            uint4 k3 = *(const uint4*)(kg + 24);
            *(uint4*)&Ks[krow][kcol]      = k0;
            *(uint4*)&Ks[krow][kcol + 8]  = k1;
            *(uint4*)&Ks[krow][kcol + 16] = k2;
            *(uint4*)&Ks[krow][kcol + 24] = k3;

            const unsigned short* vg = Vb + (size_t)vrow * 2048 + kv;
            uint4 v0 = *(const uint4*)vg;
            uint4 v1 = *(const uint4*)(vg + 8);
            uint4 v2 = *(const uint4*)(vg + 16);
            uint4 v3 = *(const uint4*)(vg + 24);
            *(uint4*)&Vs[vrow][0]  = v0;
            *(uint4*)&Vs[vrow][8]  = v1;
            *(uint4*)&Vs[vrow][16] = v2;
            *(uint4*)&Vs[vrow][24] = v3;
        }
        __syncthreads();

        // ---- S = Q @ K^T, two 16x16 column tiles ----
        v8f s0 = v8f_zero(), s1 = v8f_zero();
#pragma unroll
        for (int tile = 0; tile < 2; ++tile) {
            const int krl = tile * 16 + l15;
#pragma unroll
            for (int kk = 0; kk < 4; ++kk) {
                FragBF16 kf;
#pragma unroll
                for (int j = 0; j < 8; ++j) {
                    const int d = kk * 32 + db + 2 * j + ((j < 4) ? 0 : 8);
                    kf.u[j] = *(const unsigned int*)&Ks[krl][d];
                }
                if (tile == 0) s0 = wmma_bf16(qf[kk].v, kf.v, s0);
                else           s1 = wmma_bf16(qf[kk].v, kf.v, s1);
            }
        }

        // ---- causal mask + online softmax ----
        float p0[8], p1[8], scl[8];
#pragma unroll
        for (int r = 0; r < 8; ++r) {
            const int rowg = qbase + r + lhalf * 8;
            float v0 = s0[r]; if (kv + l15 > rowg)      v0 = -1e30f;
            float v1 = s1[r]; if (kv + 16 + l15 > rowg) v1 = -1e30f;
            float mx = fmaxf(v0, v1);
            mx = fmaxf(mx, __shfl_xor(mx, 1, 32));
            mx = fmaxf(mx, __shfl_xor(mx, 2, 32));
            mx = fmaxf(mx, __shfl_xor(mx, 4, 32));
            mx = fmaxf(mx, __shfl_xor(mx, 8, 32));
            const float mnew = fmaxf(Mrun[r], mx);
            const float sc   = __expf(Mrun[r] - mnew);
            Mrun[r] = mnew;
            const float e0 = __expf(v0 - mnew);
            const float e1 = __expf(v1 - mnew);
            float rs = e0 + e1;
            rs += __shfl_xor(rs, 1, 32);
            rs += __shfl_xor(rs, 2, 32);
            rs += __shfl_xor(rs, 4, 32);
            rs += __shfl_xor(rs, 8, 32);
            Lrun[r] = Lrun[r] * sc + rs;
            scl[r] = sc; p0[r] = e0; p1[r] = e1;
        }
#pragma unroll
        for (int dt = 0; dt < 8; ++dt)
#pragma unroll
            for (int r = 0; r < 8; ++r) o[dt][r] *= scl[r];

        // ---- P: C-layout -> A-layout via per-wave LDS bounce ----
#pragma unroll
        for (int r = 0; r < 8; ++r) {
            const int rl = r + lhalf * 8;
            Pls[wave][rl][l15]      = f2bf(p0[r]);
            Pls[wave][rl][l15 + 16] = f2bf(p1[r]);
        }
        FragBF16 pf;
#pragma unroll
        for (int j = 0; j < 8; ++j) {
            const int kc = (j < 4) ? (kb2 + 2 * j) : (kb2 + 16 + 2 * (j - 4));
            pf.u[j] = *(const unsigned int*)&Pls[wave][l15][kc];
        }

        // ---- O += P @ V  (V^T staged tile -> contiguous u32 frags) ----
#pragma unroll
        for (int dt = 0; dt < 8; ++dt) {
            FragBF16 vf;
            const int d = dt * 16 + l15;
#pragma unroll
            for (int j = 0; j < 8; ++j)
                vf.u[j] = *(const unsigned int*)&Vs[d][tb + 2 * j + ((j < 4) ? 0 : 8)];
            o[dt] = wmma_bf16(pf.v, vf.v, o[dt]);
        }
        __syncthreads();
    }

    // ---- epilogue: O / L -> attn buffer [B*T, C] bf16 ----
    const int b = bh >> 4;
    const int h = bh & 15;
#pragma unroll
    for (int dt = 0; dt < 8; ++dt) {
#pragma unroll
        for (int r = 0; r < 8; ++r) {
            const int rowg = qbase + r + lhalf * 8;
            O[((size_t)(b * 2048 + rowg)) * 2048 + h * 128 + dt * 16 + l15] =
                f2bf(o[dt][r] / Lrun[r]);
        }
    }
}

// ---------------------------------------------------------------------------
// Host-side launch
// ---------------------------------------------------------------------------
extern "C" void kernel_launch(void* const* d_in, const int* in_sizes, int n_in,
                              void* d_out, int out_size, void* d_ws, size_t ws_size,
                              hipStream_t stream) {
    const float* x      = (const float*)d_in[0];  // [4,2048,2048]
    const float* W_qkv  = (const float*)d_in[1];  // [2048,6144]
    const float* b_qkv  = (const float*)d_in[2];  // [6144]
    const float* W_proj = (const float*)d_in[3];  // [2048,2048]
    const float* b_proj = (const float*)d_in[4];  // [2048]
    float* out = (float*)d_out;                   // [4,2048,2048]

    const size_t N_X     = 4ull * 2048 * 2048;    // 16,777,216
    const size_t N_WQKV  = 2048ull * 6144;        // 12,582,912
    const size_t N_WPROJ = 2048ull * 2048;        //  4,194,304
    const size_t N_BHTD  = 4ull * 16 * 2048 * 128;// 16,777,216

    unsigned short* p      = (unsigned short*)d_ws;
    unsigned short* xb     = p; p += N_X;
    unsigned short* wqkvT  = p; p += N_WQKV;   // [6144][2048] bf16
    unsigned short* wprojT = p; p += N_WPROJ;  // [2048][2048] bf16
    unsigned short* qb     = p; p += N_BHTD;
    unsigned short* kb     = p; p += N_BHTD;
    unsigned short* vtb    = p; p += N_BHTD;
    unsigned short* attnb  = p; p += N_X;

    // Stage 0: fp32 -> bf16 (x linear; weights transposed to [N][K])
    {
        int n4 = (int)(N_X / 4);
        cvt_bf16_kernel<<<(n4 + 255) / 256, 256, 0, stream>>>(x, xb, n4);
        dim3 g1(6144 / 32, 2048 / 32);
        cvt_transpose_kernel<<<g1, 256, 0, stream>>>(W_qkv, wqkvT, 2048, 6144);
        dim3 g2(2048 / 32, 2048 / 32);
        cvt_transpose_kernel<<<g2, 256, 0, stream>>>(W_proj, wprojT, 2048, 2048);
    }

    // Stage 1: qkv = x @ W_qkv + b_qkv -> Q,K [B,H,T,D] / Vt [B,H,D,T]
    {
        dim3 grid(6144 / 256, 8192 / 128);  // (24, 64)
        gemm_bf16_kernel<<<grid, 256, 0, stream>>>(xb, wqkvT, b_qkv, 0,
                                                   qb, kb, vtb, nullptr);
    }

    // Stage 2: causal flash attention -> attnb [B*T, C]
    {
        dim3 grid(2048 / 64, 4 * 16);  // (32, 64)
        flash_attn_kernel<<<grid, 128, 0, stream>>>(qb, kb, vtb, attnb);
    }

    // Stage 3: out = attnb @ W_proj + b_proj (fp32)
    {
        dim3 grid(2048 / 256, 8192 / 128);  // (8, 64)
        gemm_bf16_kernel<<<grid, 256, 0, stream>>>(attnb, wprojT, b_proj, 1,
                                                   nullptr, nullptr, nullptr, out);
    }
}
